// KAN_Rnn_6253472383107
// MI455X (gfx1250) — matrix operand order repaired and checked by reading, not compiled, not run on hardware
//
#include <hip/hip_runtime.h>

typedef __attribute__((ext_vector_type(16))) __bf16 v16bf;
typedef __attribute__((ext_vector_type(8)))  __bf16 v8bf;
typedef __attribute__((ext_vector_type(8)))  float  v8f;

#define KC      128      // K-chunk staged in LDS (divides every region exactly)
#define HDIM    256      // hidden width
#define TSTEPS  128
#define INDIM   16

__host__ __device__ constexpr int roundup128(int x) { return (x + 127) & ~127; }

struct Params {
  const float* x_seq;
  const __bf16* Wsp[6];   // [256][8F]    spline weights (row o = flat sw[o,:,:])
  const __bf16* Wb[6];    // [256][Kbp]   base weights, Kbp = roundup128(F), zero padded
  const float* out_w;
  const float* out_b;
  float* out;
};

__device__ __forceinline__ float siluf(float x) { return x / (1.f + __expf(-x)); }

// Cox-de Boor, order 3, uniform knots t_i = -2.2 + 0.4*i (i = 0..11) -> 8 bases
__device__ __forceinline__ void bspl8(float x, float b[8]) {
  float c[11];
  #pragma unroll
  for (int i = 0; i < 11; ++i) {
    float t0 = -2.2f + 0.4f * (float)i;
    c[i] = (x >= t0 && x < t0 + 0.4f) ? 1.f : 0.f;
  }
  #pragma unroll
  for (int j = 1; j <= 3; ++j) {
    float inv = 1.f / (0.4f * (float)j);
    #pragma unroll
    for (int i = 0; i < 11 - j; ++i) {
      float ti    = -2.2f + 0.4f * (float)i;
      float tipj1 = ti + 0.4f * (float)(j + 1);
      c[i] = (x - ti) * inv * c[i] + (tipj1 - x) * inv * c[i + 1];
    }
  }
  #pragma unroll
  for (int i = 0; i < 8; ++i) b[i] = c[i];
}

// z accessor; LI is compile-time so the selection folds away
template<int LI>
__device__ __forceinline__ float zload_t(int row, int f, const float* sX,
    const __bf16* sH1, const __bf16* sH2, const __bf16* sH3, const float* sT) {
  if constexpr (LI == 0) {
    return (f < INDIM) ? sX[row*INDIM + f] : (float)sH1[row*HDIM + (f - INDIM)];
  } else if constexpr (LI == 2) {
    return (f < HDIM) ? (float)sH1[row*HDIM + f] : (float)sH2[row*HDIM + (f - HDIM)];
  } else if constexpr (LI == 4) {
    return (f < HDIM) ? (float)sH2[row*HDIM + f] : (float)sH3[row*HDIM + (f - HDIM)];
  } else {
    return sT[row*HDIM + f];   // layers 1,3,5 read layer-1 pre-activation
  }
}

// Stage one KC-wide chunk of the A panel (16 rows x KC bf16) into LDS.
template<int LI, int F, bool BASE>
__device__ __forceinline__ void stage_chunk(int k0, __bf16* __restrict__ dst, int tid,
    const float* sX, const __bf16* sH1, const __bf16* sH2, const __bf16* sH3,
    const float* sT) {
  if constexpr (!BASE) {
    // spline chunk: 16 rows x 16 features = 256 tasks, exactly one per thread
    const int f0 = k0 >> 3;
    const int fl = tid >> 4, row = tid & 15;
    float x = zload_t<LI>(row, f0 + fl, sX, sH1, sH2, sH3, sT);
    float b[8]; bspl8(x, b);
    v8bf pk;
    #pragma unroll
    for (int c = 0; c < 8; ++c) pk[c] = (__bf16)b[c];
    *(v8bf*)(dst + row*KC + fl*8) = pk;                  // ds_store_b128
  } else {
    // base chunk: silu(z) (zero padded past F), 8 scalar stores per thread
    #pragma unroll
    for (int e = tid; e < 16*KC; e += 256) {
      int j = e >> 4, row = e & 15;
      int f = k0 + j;
      float v = (f < F) ? siluf(zload_t<LI>(row, f, sX, sH1, sH2, sH3, sT)) : 0.f;
      dst[row*KC + j] = (__bf16)v;
    }
  }
}

// One weight region (spline or base) of one layer: double-buffered A staging + WMMA.
template<int LI, int F, bool BASE>
__device__ __forceinline__ void gemm_region(const __bf16* __restrict__ W,
    v8f& acc0, v8f& acc1, __bf16 (*sA)[16*KC], int& buf,
    int tid, int wave, int lane,
    const float* sX, const __bf16* sH1, const __bf16* sH2, const __bf16* sH3,
    const float* sT) {
  constexpr int sK  = BASE ? roundup128(F) : 8*F;   // row stride (elements)
  constexpr int nch = sK / KC;                      // exact by construction
  const int lc = lane & 15, hi = lane >> 4;
  // per-lane B row pointers, computed ONCE per region
  const __bf16* Wl0 = W + (size_t)(wave*32 + lc)*sK + 16*hi;
  const __bf16* Wl1 = Wl0 + (size_t)16*sK;

  stage_chunk<LI,F,BASE>(0, sA[buf], tid, sX, sH1, sH2, sH3, sT);
  __syncthreads();
  for (int ci = 0; ci < nch; ++ci) {
    if (ci + 1 < nch)
      stage_chunk<LI,F,BASE>((ci+1)*KC, sA[buf^1], tid, sX, sH1, sH2, sH3, sT);
    __syncthreads();
    const __bf16* __restrict__ sAb = sA[buf] + lc*KC + 16*hi;
    const __bf16* __restrict__ w0  = Wl0 + ci*KC;
    const __bf16* __restrict__ w1  = Wl1 + ci*KC;
    // pull the next K-panel of this weight matrix toward the near caches
    __builtin_prefetch((const void*)(w0 + KC), 0, 3);
    __builtin_prefetch((const void*)(w1 + KC), 0, 3);
    #pragma unroll
    for (int kk = 0; kk < KC; kk += 32) {
      // A fragment: lane (hi,lc) = row lc, K = kk+16hi..+15
      union { v16bf v; v8bf h[2]; } a;
      a.h[0] = *(const v8bf*)(sAb + kk);               // ds_load_b128
      a.h[1] = *(const v8bf*)(sAb + kk + 8);           // ds_load_b128
      // B fragments: lane holds 16 contiguous K of one N column (2x b128)
      v16bf b0 = *(const v16bf*)(w0 + kk);
      v16bf b1 = *(const v16bf*)(w1 + kk);
      acc0 = __builtin_amdgcn_wmma_f32_16x16x32_bf16(false, a.v, false, b0,
                                                     (short)0, acc0, false, false);
      acc1 = __builtin_amdgcn_wmma_f32_16x16x32_bf16(false, a.v, false, b1,
                                                     (short)0, acc1, false, false);
    }
    buf ^= 1;
    __syncthreads();
  }
}

// Full KAN layer: base + spline accumulate into the same WMMA accumulators.
template<int LI, int F>
__device__ __forceinline__ void kan_layer(const Params& p, __bf16 (*sA)[16*KC], int& buf,
    int tid, int wave, int lane,
    const float* sX, __bf16* sH1, __bf16* sH2, __bf16* sH3, float* sT) {
  v8f acc0 = {}, acc1 = {};
  gemm_region<LI,F,false>(p.Wsp[LI], acc0, acc1, sA, buf, tid, wave, lane,
                          sX, sH1, sH2, sH3, sT);
  gemm_region<LI,F,true >(p.Wb[LI],  acc0, acc1, sA, buf, tid, wave, lane,
                          sX, sH1, sH2, sH3, sT);
  // D layout: lane l, vgpr r -> M = r + 8*(l>>4), N = l&15
  const int lc = lane & 15, hi = lane >> 4;
  const int colBase = wave*32 + lc;
  if constexpr (LI & 1) {          // layer 2 of a cell: tanh -> new recurrent state
    __bf16* dstH = (LI == 1) ? sH1 : (LI == 3) ? sH2 : sH3;
    #pragma unroll
    for (int r = 0; r < 8; ++r) {
      int row = r + 8*hi;
      dstH[row*HDIM + colBase]      = (__bf16)tanhf(acc0[r]);
      dstH[row*HDIM + colBase + 16] = (__bf16)tanhf(acc1[r]);
    }
  } else {                         // layer 1: raw pre-activation feeds layer 2
    #pragma unroll
    for (int r = 0; r < 8; ++r) {
      int row = r + 8*hi;
      sT[row*HDIM + colBase]      = acc0[r];
      sT[row*HDIM + colBase + 16] = acc1[r];
    }
  }
  __syncthreads();
}

__global__ void __launch_bounds__(256)
kan_rnn_kernel(Params p) {
  // LDS: 8 + 24 + 16 + 1 = 49 KB
  __shared__ alignas(32) __bf16 sA[2][16 * KC];
  __shared__ alignas(32) __bf16 sH1[16 * HDIM];
  __shared__ alignas(32) __bf16 sH2[16 * HDIM];
  __shared__ alignas(32) __bf16 sH3[16 * HDIM];
  __shared__ float  sT[16 * HDIM];
  __shared__ float  sX[16 * INDIM];

  const int tid  = threadIdx.x;
  const int lane = tid & 31;
  const int wave = tid >> 5;
  const int b0   = blockIdx.x * 16;   // 8 blocks x 16 batch rows
  int buf = 0;

  for (int i = tid; i < 16 * HDIM; i += 256) {
    sH1[i] = (__bf16)0.f; sH2[i] = (__bf16)0.f; sH3[i] = (__bf16)0.f;
  }
  __syncthreads();

  for (int t = 0; t < TSTEPS; ++t) {
    { int row = tid >> 4, f = tid & 15;
      sX[tid] = p.x_seq[((size_t)(b0 + row) * TSTEPS + t) * INDIM + f]; }
    __syncthreads();

    kan_layer<0,272>(p, sA, buf, tid, wave, lane, sX, sH1, sH2, sH3, sT);
    kan_layer<1,256>(p, sA, buf, tid, wave, lane, sX, sH1, sH2, sH3, sT);
    kan_layer<2,512>(p, sA, buf, tid, wave, lane, sX, sH1, sH2, sH3, sT);
    kan_layer<3,256>(p, sA, buf, tid, wave, lane, sX, sH1, sH2, sH3, sT);
    kan_layer<4,512>(p, sA, buf, tid, wave, lane, sX, sH1, sH2, sH3, sT);
    kan_layer<5,256>(p, sA, buf, tid, wave, lane, sX, sH1, sH2, sH3, sT);
  }

  // final projection: out[b][j] = h3[b] . out_w[j] + out_b[j]  (16 rows x 3)
  if (tid < 48) {
    int row = tid / 3, j = tid - row * 3;
    float s = p.out_b[j];
    for (int o = 0; o < HDIM; ++o)
      s += (float)sH3[row*HDIM + o] * p.out_w[j*HDIM + o];
    p.out[(size_t)(b0 + row) * 3 + j] = s;
  }
}

// ---------- weight preprocessing: fp32 -> bf16 panels in workspace ----------
__global__ void cvt_copy_kernel(const float* __restrict__ src, __bf16* __restrict__ dst, int n) {
  int i = blockIdx.x * 256 + threadIdx.x;
  if (i < n) dst[i] = (__bf16)src[i];
}
__global__ void cvt_pad_kernel(const float* __restrict__ src, __bf16* __restrict__ dst,
                               int F, int Kp, int n) {
  int i = blockIdx.x * 256 + threadIdx.x;
  if (i < n) {
    int o = i / Kp, f = i - o * Kp;
    dst[i] = (f < F) ? (__bf16)src[o * F + f] : (__bf16)0.f;
  }
}

extern "C" void kernel_launch(void* const* d_in, const int* in_sizes, int n_in,
                              void* d_out, int out_size, void* d_ws, size_t ws_size,
                              hipStream_t stream) {
  (void)in_sizes; (void)n_in; (void)out_size; (void)ws_size;
  const int Fs[6]    = {272, 256, 512, 256, 512, 256};
  const int bwIdx[6] = {1, 3, 5, 7, 9, 11};
  const int swIdx[6] = {2, 4, 6, 8, 10, 12};

  __bf16* ws = (__bf16*)d_ws;
  Params p{};
  p.x_seq = (const float*)d_in[0];
  p.out_w = (const float*)d_in[13];
  p.out_b = (const float*)d_in[14];
  p.out   = (float*)d_out;

  size_t off = 0;   // bf16 elements; total ~4.78M elements = 9.1 MB of d_ws
  for (int li = 0; li < 6; ++li) {
    int F = Fs[li];
    int Ks = 8 * F;
    int Kbp = roundup128(F);
    __bf16* Wsp = ws + off; off += (size_t)256 * Ks;
    __bf16* Wb  = ws + off; off += (size_t)256 * Kbp;
    int nsp = 256 * Ks, nb = 256 * Kbp;
    cvt_copy_kernel<<<(nsp + 255) / 256, 256, 0, stream>>>(
        (const float*)d_in[swIdx[li]], Wsp, nsp);
    cvt_pad_kernel<<<(nb + 255) / 256, 256, 0, stream>>>(
        (const float*)d_in[bwIdx[li]], Wb, F, Kbp, nb);
    p.Wsp[li] = Wsp;
    p.Wb[li]  = Wb;
  }

  kan_rnn_kernel<<<dim3(8), dim3(256), 0, stream>>>(p);
}
